// LiftSplatShoot_17781164605693
// MI455X (gfx1250) — matrix operand order repaired and checked by reading, not compiled, not run on hardware
//
#include <hip/hip_runtime.h>
#include <hip/hip_bf16.h>

typedef __attribute__((ext_vector_type(16))) __bf16 bf16x16;
typedef __attribute__((ext_vector_type(8)))  float  f32x8;

namespace lss {
constexpr int BN   = 6;          // B*N cameras
constexpr int INC  = 256;        // input channels
constexpr int D    = 41;         // depth bins
constexpr int C    = 64;         // cam feature channels
constexpr int DC   = 105;        // D + C
constexpr int FH   = 28;
constexpr int FW   = 50;
constexpr int HW   = 1400;       // FH*FW
constexpr int NP   = BN * D * HW;   // 344400 lifted points
constexpr int GX   = 33, GY = 33, GZ = 2;
constexpr int PIX  = GX * GY;    // 1089
constexpr int CZ   = C * GZ;     // 128 BEV channels
constexpr int NTP  = 69;         // ceil(1089/16) pixel tiles for convs
constexpr float BNSCALE = 0.99999500003749981f; // 1/sqrt(1+1e-5), eval-mode BN

__device__ __forceinline__ f32x8 wmma_bf16(const bf16x16& a, const bf16x16& b,
                                           const f32x8& c) {
  // (neg_a, A, neg_b, B, c_mod, C, reuse_a, reuse_b)
  return __builtin_amdgcn_wmma_f32_16x16x32_bf16(false, a, false, b, (short)0, c,
                                                 false, false);
}
} // namespace lss

// ===========================================================================
// Panel layout (both A and B):  panel[((tile*KS + ks)*32 + lane)*16 + j]
//   A element j of lane: K = ks*32 + 16*(lane>>4) + j,               M row = tile*16 + (lane&15)
//   B element j of lane: K = ks*32 + 16*(lane>>4) + j,               N col = tile*16 + (lane&15)
// GEMM inner loop then reads 32 contiguous bytes per lane per operand.
// ===========================================================================

// ---- pack A: dense row-major (Mrows x Kdim), zero-padded ------------------
__global__ void lss_pack_a_dense(const float* __restrict__ A, __bf16* __restrict__ apan,
                                 int Mrows, int Kdim, int KS) {
  int tid = blockIdx.x * blockDim.x + threadIdx.x;
  int j    = tid & 15;
  int lane = (tid >> 4) & 31;
  int t    = tid >> 9;          // mt*KS + ks
  int ks   = t % KS;
  int mt   = t / KS;
  int hf = lane >> 4, lr = lane & 15;
  int mrow = mt * 16 + lr;
  int k = ks * 32 + (hf << 4) + j;
  float v = (mrow < Mrows) ? A[mrow * Kdim + k] : 0.0f;
  apan[tid] = (__bf16)v;
}

// ---- pack A: conv weights OIHW, K ordered as k = r*Ci + ci ----------------
__global__ void lss_pack_a_conv(const float* __restrict__ wgt, __bf16* __restrict__ apan,
                                int Ci, int ciShift, int KS) {
  int tid = blockIdx.x * blockDim.x + threadIdx.x;
  int j    = tid & 15;
  int lane = (tid >> 4) & 31;
  int t    = tid >> 9;
  int ks   = t % KS;
  int mt   = t / KS;
  int hf = lane >> 4, lr = lane & 15;
  int m  = mt * 16 + lr;                       // always < Co
  int ka = ks * 32 + (hf << 4) + j;
  int r  = ka >> ciShift;
  int ci = ka & (Ci - 1);
  apan[tid] = (__bf16)wgt[(m * Ci + ci) * 9 + r];
}

// ---- pack B: depthnet activations x (6 images, 256 x 1400 each) -----------
__global__ void lss_pack_b_x(const float* __restrict__ x, __bf16* __restrict__ bpan) {
  using namespace lss;
  int tid = blockIdx.x * blockDim.x + threadIdx.x;
  int j    = tid & 15;
  int lane = (tid >> 4) & 31;
  int t    = tid >> 9;          // (n*88 + nt)*8 + ks
  int ks   = t & 7;
  int u    = t >> 3;
  int nt   = u % 88;
  int n    = u / 88;
  int hf = lane >> 4, lr = lane & 15;
  int col = nt * 16 + lr;
  int k   = ks * 32 + (hf << 4) + j;
  float v = (col < HW) ? x[(n * INC + k) * HW + col] : 0.0f;
  bpan[tid] = (__bf16)v;
}

// ---- pack B: im2col of a (Ci x 33 x 33) activation, SAME zero padding -----
__global__ void lss_pack_b_conv(const float* __restrict__ in, __bf16* __restrict__ bpan,
                                int Ci, int ciShift, int KS) {
  using namespace lss;
  int tid = blockIdx.x * blockDim.x + threadIdx.x;
  int j    = tid & 15;
  int lane = (tid >> 4) & 31;
  int t    = tid >> 9;          // nt*KS + ks
  int ks   = t % KS;
  int nt   = t / KS;
  int hf = lane >> 4, lr = lane & 15;
  int npix = nt * 16 + lr;
  int kb  = ks * 32 + (hf << 4) + j;
  int rb  = kb >> ciShift;
  int cib = kb & (Ci - 1);
  int ky = rb / 3, kx = rb - ky * 3;
  int yy = npix / GY, xx = npix - yy * GY;
  int iy = yy + ky - 1, ix = xx + kx - 1;
  float v = 0.0f;
  if (npix < PIX && iy >= 0 && iy < GX && ix >= 0 && ix < GY)
    v = in[cib * PIX + iy * GY + ix];
  bpan[tid] = (__bf16)v;
}

// ---------------------------------------------------------------------------
// depthnet GEMM: feat(n, 0:105, hw) = Wd @ x + bd.  One 16x16 tile per wave,
// branch-free inner loop over packed panels, unrolled x2.
// ---------------------------------------------------------------------------
__global__ void lss_depthnet_gemm(const __bf16* __restrict__ apan,
                                  const __bf16* __restrict__ bpan,
                                  const float* __restrict__ bd,
                                  float* __restrict__ feat) {
  using namespace lss;
  constexpr int NT = 88, MT = 7, KS = 8;
  int wave = (blockIdx.x * blockDim.x + threadIdx.x) >> 5;
  int lane = threadIdx.x & 31;
  if (wave >= BN * MT * NT) return;
  int nt = wave % NT;
  int mt = (wave / NT) % MT;
  int n  = wave / (NT * MT);
  int hf = lane >> 4, lr = lane & 15;
  const bf16x16* ap = (const bf16x16*)apan + (mt * KS) * 32 + lane;
  const bf16x16* bp = (const bf16x16*)bpan + ((n * NT + nt) * KS) * 32 + lane;
  f32x8 acc = {};
#pragma unroll
  for (int ks = 0; ks < KS; ks += 2) {
    bf16x16 a0 = ap[(ks + 0) * 32];
    bf16x16 b0 = bp[(ks + 0) * 32];
    bf16x16 a1 = ap[(ks + 1) * 32];
    bf16x16 b1 = bp[(ks + 1) * 32];
    acc = wmma_bf16(a0, b0, acc);
    acc = wmma_bf16(a1, b1, acc);
  }
  int col = nt * 16 + lr;
#pragma unroll
  for (int i = 0; i < 8; ++i) {
    int m = mt * 16 + (hf << 3) + i;
    if (m < DC && col < HW) feat[(n * DC + m) * HW + col] = acc[i] + bd[m];
  }
}

// ---------------------------------------------------------------------------
// conv GEMM: out(Co,1089) = Wpanel @ im2col-panel, fused BN+ReLU epilogue.
// ---------------------------------------------------------------------------
__global__ void lss_conv_gemm(const __bf16* __restrict__ apan,
                              const __bf16* __restrict__ bpan,
                              float* __restrict__ out,
                              int MT, int KS) {
  using namespace lss;
  int wave = (blockIdx.x * blockDim.x + threadIdx.x) >> 5;
  int lane = threadIdx.x & 31;
  if (wave >= MT * NTP) return;
  int nt = wave % NTP;
  int mt = wave / NTP;
  int hf = lane >> 4, lr = lane & 15;
  const bf16x16* ap = (const bf16x16*)apan + (mt * KS) * 32 + lane;
  const bf16x16* bp = (const bf16x16*)bpan + (nt * KS) * 32 + lane;
  f32x8 acc = {};
  for (int ks = 0; ks < KS; ks += 2) {   // KS is 36 or 144: always even
    bf16x16 a0 = ap[(ks + 0) * 32];
    bf16x16 b0 = bp[(ks + 0) * 32];
    bf16x16 a1 = ap[(ks + 1) * 32];
    bf16x16 b1 = bp[(ks + 1) * 32];
    acc = wmma_bf16(a0, b0, acc);
    acc = wmma_bf16(a1, b1, acc);
  }
  int npix = nt * 16 + lr;
#pragma unroll
  for (int i = 0; i < 8; ++i) {
    int mo = mt * 16 + (hf << 3) + i;
    if (npix < PIX)
      out[mo * PIX + npix] = fmaxf(acc[i] * BNSCALE, 0.0f);
  }
}

// ---------------------------------------------------------------------------
// softmax over depth bins + split context channels (one thread / column)
// ---------------------------------------------------------------------------
__global__ void lss_softmax_split(const float* __restrict__ feat,
                                  float* __restrict__ depth,
                                  float* __restrict__ ctx) {
  using namespace lss;
  int t = blockIdx.x * blockDim.x + threadIdx.x;
  if (t >= BN * HW) return;
  int n = t / HW, hw = t % HW;
  const float* f = feat + n * DC * HW + hw;
  float mx = -3.0e38f;
  for (int d = 0; d < D; ++d) mx = fmaxf(mx, f[d * HW]);
  float s = 0.0f;
  for (int d = 0; d < D; ++d) s += __expf(f[d * HW] - mx);
  float inv = 1.0f / s;
  for (int d = 0; d < D; ++d)
    depth[(n * D + d) * HW + hw] = __expf(f[d * HW] - mx) * inv;
  for (int c = 0; c < C; ++c)
    ctx[(n * C + c) * HW + hw] = f[(D + c) * HW];
}

__global__ void lss_zero(float* __restrict__ p, int nfl) {
  for (int i = blockIdx.x * blockDim.x + threadIdx.x; i < nfl;
       i += gridDim.x * blockDim.x) p[i] = 0.0f;
}

// ---------------------------------------------------------------------------
// geometry + voxel scatter: grid0[(z*64+c)*1089 + x*33 + y] += d * ctx[c]
// ---------------------------------------------------------------------------
__global__ void lss_scatter(const float* __restrict__ depth,
                            const float* __restrict__ ctx,
                            const float* __restrict__ rots,
                            const float* __restrict__ trans,
                            float* __restrict__ grid0) {
  using namespace lss;
  __shared__ float s_dv[4];
  __shared__ int   s_vox[4];
  int py = threadIdx.y;
  int p  = blockIdx.x * 4 + py;
  if (threadIdx.x == 0) {
    int vox = -1; float dv = 0.0f;
    if (p < NP) {
      int w  = p % FW;
      int h  = (p / FW) % FH;
      int nd = p / HW;          // n*D + d
      int d  = nd % D;
      int n  = nd / D;
      float dep = 4.0f + (float)d;
      float u = (float)w * (799.0f / 49.0f);   // linspace(0, OGW-1, FW)
      float v = (float)h * (447.0f / 27.0f);   // linspace(0, OGH-1, FH)
      float px = u * dep, pv = v * dep, pz = dep;
      const float* r = rots + n * 9;
      float gx = r[0]*px + r[1]*pv + r[2]*pz + trans[n*3+0];
      float gy = r[3]*px + r[4]*pv + r[5]*pz + trans[n*3+1];
      float gz = r[6]*px + r[7]*pv + r[8]*pz + trans[n*3+2];
      int ix = (int)((gx + 50.0f) / 3.0f);     // trunc-to-zero == jnp cast
      int iy = (int)((gy + 50.0f) / 3.0f);
      int iz = (int)((gz + 5.0f)  / 3.0f);
      if (ix >= 0 && ix < GX && iy >= 0 && iy < GY && iz >= 0 && iz < GZ) {
        vox = iz * (C * PIX) + ix * GY + iy;
        dv  = depth[p];
      }
    }
    s_vox[py] = vox; s_dv[py] = dv;
  }
  __syncthreads();
  int vox = s_vox[py];
  if (p < NP && vox >= 0) {
    int c  = threadIdx.x;
    int n  = p / (D * HW);
    int hw = p % HW;
    atomicAdd(grid0 + vox + c * PIX, s_dv[py] * ctx[(n * C + c) * HW + hw]);
  }
}

// ---------------------------------------------------------------------------
// launcher
// ---------------------------------------------------------------------------
extern "C" void kernel_launch(void* const* d_in, const int* in_sizes, int n_in,
                              void* d_out, int out_size, void* d_ws, size_t ws_size,
                              hipStream_t stream) {
  using namespace lss;
  const float* x     = (const float*)d_in[0];   // (1,6,256,28,50)
  const float* rots  = (const float*)d_in[1];   // (1,6,3,3)
  const float* trans = (const float*)d_in[2];   // (1,6,3)
  const float* Wd    = (const float*)d_in[3];   // (105,256,1,1)
  const float* bd    = (const float*)d_in[4];   // (105)
  const float* W1    = (const float*)d_in[5];   // (128,128,3,3)
  const float* W2    = (const float*)d_in[6];   // (512,128,3,3)
  const float* W3    = (const float*)d_in[7];   // (512,512,3,3)
  const float* W4    = (const float*)d_in[8];   // (256,512,3,3)
  float* out = (float*)d_out;                   // (1,256,33,33)

  size_t off = 0;
  char* base = (char*)d_ws;
  auto alloc = [&](size_t bytes) -> char* {
    char* p = base + off;
    off += (bytes + 255) & ~(size_t)255;
    return p;
  };
  float*  feat  = (float*)alloc((size_t)BN * DC * HW * 4);   // 3.5 MB
  float*  depth = (float*)alloc((size_t)BN * D  * HW * 4);   // 1.4 MB
  float*  ctx   = (float*)alloc((size_t)BN * C  * HW * 4);   // 2.2 MB
  float*  grid0 = (float*)alloc((size_t)CZ * PIX * 4);
  float*  h1    = (float*)alloc((size_t)CZ * PIX * 4);
  float*  h2    = (float*)alloc((size_t)512 * PIX * 4);
  float*  h3    = (float*)alloc((size_t)512 * PIX * 4);
  __bf16* apan  = (__bf16*)alloc((size_t)32 * 144 * 512 * 2); // max A panel (conv3)
  __bf16* bpan  = (__bf16*)alloc((size_t)BN * 88 * 8 * 512 * 2 > (size_t)NTP * 144 * 512 * 2
                                 ? (size_t)NTP * 144 * 512 * 2
                                 : (size_t)NTP * 144 * 512 * 2); // 10.2 MB >= depthnet's 4.3 MB

  auto cdiv = [](int a, int b) { return (a + b - 1) / b; };

  // ---- depthnet GEMM (pack + WMMA) ----
  lss_pack_a_dense<<<cdiv(7 * 8 * 512, 256), 256, 0, stream>>>(Wd, apan, DC, INC, 8);
  lss_pack_b_x   <<<cdiv(BN * 88 * 8 * 512, 256), 256, 0, stream>>>(x, bpan);
  lss_depthnet_gemm<<<cdiv(BN * 7 * 88, 8), 256, 0, stream>>>(apan, bpan, bd, feat);

  // ---- lift + splat ----
  lss_softmax_split<<<cdiv(BN * HW, 128), 128, 0, stream>>>(feat, depth, ctx);
  lss_zero<<<272, 256, 0, stream>>>(grid0, CZ * PIX);
  dim3 sb(64, 4);
  lss_scatter<<<cdiv(NP, 4), sb, 0, stream>>>(depth, ctx, rots, trans, grid0);

  // ---- BEV conv stack (pack + WMMA per layer, panels reused) ----
  auto conv = [&](const float* in, const float* w, float* o, int Ci, int Co, int sh) {
    int KS = (Ci * 9) / 32;      // 36 or 144
    int MT = Co / 16;
    lss_pack_a_conv<<<cdiv(MT * KS * 512, 256), 256, 0, stream>>>(w, apan, Ci, sh, KS);
    lss_pack_b_conv<<<cdiv(NTP * KS * 512, 256), 256, 0, stream>>>(in, bpan, Ci, sh, KS);
    lss_conv_gemm<<<cdiv(MT * NTP, 8), 256, 0, stream>>>(apan, bpan, o, MT, KS);
  };
  conv(grid0, W1, h1,  128, 128, 7);
  conv(h1,    W2, h2,  128, 512, 7);
  conv(h2,    W3, h3,  512, 512, 9);
  conv(h3,    W4, out, 512, 256, 9);
}
